// TriangleMultiplicativeUpdate_57561151701395
// MI455X (gfx1250) — matrix operand order repaired and checked by reading, not compiled, not run on hardware
//
#include <hip/hip_runtime.h>
#include <hip/hip_bf16.h>

#define NDIM 512
#define NN   (512 * 512)
#define CDIM 128

typedef __attribute__((ext_vector_type(16))) __bf16 v16bf;
typedef __attribute__((ext_vector_type(8)))  __bf16 v8bf;
typedef __attribute__((ext_vector_type(8)))  float  v8f;

static __device__ __forceinline__ float sigmoidf_(float x) {
    return 1.0f / (1.0f + __expf(-x));
}

// ---------------------------------------------------------------------------
// CDNA5 async global->LDS copy (16 bytes per lane), tracked by ASYNCcnt.
// lds_off is the byte offset within the workgroup's LDS allocation (this
// kernel uses only dynamic LDS, so offset 0 == start of smem[]).
static __device__ __forceinline__ void async_copy16(unsigned lds_off, const void* gptr) {
    asm volatile("global_load_async_to_lds_b128 %0, %1, off"
                 :: "v"(lds_off), "v"((unsigned long long)gptr)
                 : "memory");
}

static __device__ __forceinline__ void wait_async0() {
    asm volatile("s_wait_asynccnt 0x0" ::: "memory");
}

// ---------------------------------------------------------------------------
// WMMA bf16 fragment loaders (CDNA5 ISA 7.12.2 layouts, wave32)
// A operand: 16x32 (MxK). lane m = lane&15 -> row M; h = lane>>4 selects the
// K-octets: VGPR0..3 hold K = h*8+0..7, VGPR4..7 hold K = 16+h*8+0..7.
static __device__ __forceinline__ v16bf load_fragA(const __bf16* base, int ld, int lane) {
    const int m = lane & 15, h = lane >> 4;
    const __bf16* r = base + m * ld;
    v8bf c0 = *(const v8bf*)(r + h * 8);
    v8bf c1 = *(const v8bf*)(r + 16 + h * 8);
    v16bf f;
#pragma unroll
    for (int e = 0; e < 8; ++e) { f[e] = c0[e]; f[8 + e] = c1[e]; }
    return f;
}

// B operand: 32x16 (KxN) used for D = A * B^T with B-source stored row-major
// [N rows x K cols]. lane n = lane&15 -> column N (a row of the source);
// h = lane>>4 selects K-octets: elems 0..7 hold K = h*8+0..7, 8..15 hold K = 16+h*8+0..7.
static __device__ __forceinline__ v16bf load_fragB(const __bf16* base, int ld, int lane) {
    const int n = lane & 15, h = lane >> 4;
    const __bf16* r = base + n * ld;
    v8bf c0 = *(const v8bf*)(r + h * 8);
    v8bf c1 = *(const v8bf*)(r + 16 + h * 8);
    v16bf f;
#pragma unroll
    for (int e = 0; e < 8; ++e) { f[e] = c0[e]; f[8 + e] = c1[e]; }
    return f;
}

// 4 output tiles (one row-block rb, 4 col-blocks from cb0) of a
// [16*4 x 128] @ [128 x 128]^T GEMM, K = 128, bf16 in / f32 accum.
static __device__ __forceinline__ void gemm4(const __bf16* an, const __bf16* wmat,
                                             int rb, int cb0, int lane, v8f* acc) {
    const v8f vzero = {};
#pragma unroll
    for (int c = 0; c < 4; ++c) acc[c] = vzero;
#pragma unroll
    for (int k = 0; k < 4; ++k) {
        v16bf af = load_fragA(an + rb * 16 * CDIM + k * 32, CDIM, lane);
#pragma unroll
        for (int c = 0; c < 4; ++c) {
            v16bf bfr = load_fragB(wmat + (cb0 + c) * 16 * CDIM + k * 32, CDIM, lane);
            acc[c] = __builtin_amdgcn_wmma_f32_16x16x32_bf16(
                false, af, false, bfr, (short)0, acc[c], false, false);
        }
    }
}

// ---------------------------------------------------------------------------
// Kernel 0: convert the six 128x128 fp32 weight matrices to bf16 in workspace.
__global__ void __launch_bounds__(256)
cvt_kernel(const float* __restrict__ s0, const float* __restrict__ s1,
           const float* __restrict__ s2, const float* __restrict__ s3,
           const float* __restrict__ s4, const float* __restrict__ s5,
           __bf16* __restrict__ wb) {
    int i = blockIdx.x * 256 + threadIdx.x;
    if (i >= 6 * 16384) return;
    const float* srcs[6] = { s0, s1, s2, s3, s4, s5 };
    wb[i] = (__bf16)srcs[i >> 14][i & 16383];
}

// ---------------------------------------------------------------------------
// Kernel 1: fused LN(in) + 5 projections + sigmoid gating + mask.
// Writes a,b channel-major [C][N][N] bf16 (for the einsum) and g [P][C] bf16.
__global__ void __launch_bounds__(256)
proj_kernel(const float* __restrict__ z, const float* __restrict__ mask,
            const float* __restrict__ ln_w, const float* __restrict__ ln_b,
            const __bf16* __restrict__ wb,
            const float* __restrict__ b_ap, const float* __restrict__ b_ag,
            const float* __restrict__ b_bp, const float* __restrict__ b_bg,
            const float* __restrict__ b_g,
            __bf16* __restrict__ a_t, __bf16* __restrict__ b_t,
            __bf16* __restrict__ g_nat) {
    __shared__ float  xs[64 * 129];     // padded: bank-conflict-free row reads
    __shared__ __bf16 zn[64 * CDIM];

    const int tid = threadIdx.x;
    const int p0  = blockIdx.x * 64;

    // Coalesced cooperative load of 64 pixels x 128 channels.
    const float* zsrc = z + (size_t)p0 * CDIM;
#pragma unroll
    for (int it = 0; it < 32; ++it) {
        int flat = it * 256 + tid;                   // 0..8191
        int pl = flat >> 7, cc = flat & 127;
        xs[pl * 129 + cc] = zsrc[flat];
    }
    __syncthreads();

    // Per-pixel LayerNorm over 128 channels -> bf16.
    if (tid < 64) {
        const float* row = xs + tid * 129;
        float s = 0.f, ss = 0.f;
#pragma unroll 8
        for (int c = 0; c < CDIM; ++c) { float v = row[c]; s += v; ss += v * v; }
        float mu  = s * (1.0f / CDIM);
        float var = ss * (1.0f / CDIM) - mu * mu;
        float rs  = rsqrtf(var + 1e-5f);
#pragma unroll 8
        for (int c = 0; c < CDIM; ++c) {
            float v = (row[c] - mu) * rs * ln_w[c] + ln_b[c];
            zn[tid * CDIM + c] = (__bf16)v;
        }
    }
    __syncthreads();

    const int wave = tid >> 5, lane = tid & 31;
    const int rb = wave & 3, cb0 = (wave >> 2) * 4;
    const int h = lane >> 4, n = lane & 15;
    const int pbase = p0 + rb * 16 + 8 * h;          // C/D row M = r + 8*h

    float mk[8];
    {
        float4 m0 = *(const float4*)(mask + pbase);
        float4 m1 = *(const float4*)(mask + pbase + 4);
        mk[0] = m0.x; mk[1] = m0.y; mk[2] = m0.z; mk[3] = m0.w;
        mk[4] = m1.x; mk[5] = m1.y; mk[6] = m1.z; mk[7] = m1.w;
    }

    v8f accP[4], accG[4];

    // ---- a = (zn@w_ap^T + b_ap) * sigmoid(zn@w_ag^T + b_ag) * mask
    gemm4(zn, wb + 0 * 16384, rb, cb0, lane, accP);
    gemm4(zn, wb + 1 * 16384, rb, cb0, lane, accG);
#pragma unroll
    for (int c = 0; c < 4; ++c) {
        int o = (cb0 + c) * 16 + n;
        float bp = b_ap[o], bg = b_ag[o];
        v8bf outv;
#pragma unroll
        for (int r = 0; r < 8; ++r) {
            float val = (accP[c][r] + bp) * sigmoidf_(accG[c][r] + bg) * mk[r];
            outv[r] = (__bf16)val;
        }
        *(v8bf*)(a_t + (size_t)o * NN + pbase) = outv;   // 8 contiguous pixels
    }

    // ---- b = (zn@w_bp^T + b_bp) * sigmoid(zn@w_bg^T + b_bg) * mask
    gemm4(zn, wb + 2 * 16384, rb, cb0, lane, accP);
    gemm4(zn, wb + 3 * 16384, rb, cb0, lane, accG);
#pragma unroll
    for (int c = 0; c < 4; ++c) {
        int o = (cb0 + c) * 16 + n;
        float bp = b_bp[o], bg = b_bg[o];
        v8bf outv;
#pragma unroll
        for (int r = 0; r < 8; ++r) {
            float val = (accP[c][r] + bp) * sigmoidf_(accG[c][r] + bg) * mk[r];
            outv[r] = (__bf16)val;
        }
        *(v8bf*)(b_t + (size_t)o * NN + pbase) = outv;
    }

    // ---- g = sigmoid(zn@w_g^T + b_g), natural [P][C] layout
    gemm4(zn, wb + 4 * 16384, rb, cb0, lane, accG);
#pragma unroll
    for (int c = 0; c < 4; ++c) {
        int o = (cb0 + c) * 16 + n;
        float bg = b_g[o];
#pragma unroll
        for (int r = 0; r < 8; ++r)
            g_nat[(size_t)(pbase + r) * CDIM + o] = (__bf16)sigmoidf_(accG[c][r] + bg);
    }
}

// ---------------------------------------------------------------------------
// Kernel 2: x[c][i][j] = sum_k a[c][i][k] * b[c][j][k] (A·B^T per channel).
// Grid: (i-tiles=8, j-tiles=8, channels=128), 64x64 tile per WG, 8 waves.
// K is pipelined through a double-buffered 16 KB LDS stage filled with
// GLOBAL_LOAD_ASYNC_TO_LDS_B128 (ASYNCcnt), overlapping copies with WMMA.
__global__ void __launch_bounds__(256)
einsum_kernel(const __bf16* __restrict__ a_t, const __bf16* __restrict__ b_t,
              float* __restrict__ x_t) {
    extern __shared__ char smem[];   // 16 KB = 2 stages x (A 4KB + B 4KB)

    const int c  = blockIdx.z;
    const int i0 = blockIdx.x * 64, j0 = blockIdx.y * 64;
    const int tid = threadIdx.x, wave = tid >> 5, lane = tid & 31;
    const int rb = wave & 3;          // i sub-block (16 rows)
    const int jh = wave >> 2;         // j sub-half  (32 cols = 2 tiles)

    // Stage copy: each thread moves one 16B chunk of A and one of B per stage.
    // A/B stage tiles are 64 rows x 32 K-cols bf16, row stride 64B in LDS.
    const int row = tid >> 2, sub = tid & 3;
    const __bf16* gA = a_t + (size_t)c * NN + (size_t)(i0 + row) * NDIM + sub * 8;
    const __bf16* gB = b_t + (size_t)c * NN + (size_t)(j0 + row) * NDIM + sub * 8;
    const unsigned ldsA = (unsigned)(row * 64 + sub * 16);
    const unsigned ldsB = 4096u + (unsigned)(row * 64 + sub * 16);

    // Prologue: stage 0 into buffer 0.
    async_copy16(ldsA, gA);
    async_copy16(ldsB, gB);

    v8f acc0 = {}, acc1 = {};
    for (int s = 0; s < 16; ++s) {
        wait_async0();                 // this wave's stage-s chunks are in LDS
        __syncthreads();               // everyone's stage-s chunks are in LDS
        if (s < 15) {                  // overlap next stage with compute
            unsigned boff = (unsigned)(((s + 1) & 1) * 8192);
            async_copy16(ldsA + boff, gA + (s + 1) * 32);
            async_copy16(ldsB + boff, gB + (s + 1) * 32);
        }
        const __bf16* As = (const __bf16*)(smem + (s & 1) * 8192);
        const __bf16* Bs = (const __bf16*)(smem + 4096 + (s & 1) * 8192);
        v16bf af  = load_fragA(As + rb * 16 * 32, 32, lane);
        v16bf bf0 = load_fragB(Bs + (jh * 32) * 32,      32, lane);
        v16bf bf1 = load_fragB(Bs + (jh * 32 + 16) * 32, 32, lane);
        acc0 = __builtin_amdgcn_wmma_f32_16x16x32_bf16(false, af, false, bf0,
                                                       (short)0, acc0, false, false);
        acc1 = __builtin_amdgcn_wmma_f32_16x16x32_bf16(false, af, false, bf1,
                                                       (short)0, acc1, false, false);
    }

    const int h = lane >> 4, n = lane & 15;
    const int ibase = i0 + rb * 16 + 8 * h;
    float* xr = x_t + (size_t)c * NN;
#pragma unroll
    for (int r = 0; r < 8; ++r) {
        size_t rowoff = (size_t)(ibase + r) * NDIM + j0 + jh * 32 + n;
        xr[rowoff]      = acc0[r];
        xr[rowoff + 16] = acc1[r];
    }
}

// ---------------------------------------------------------------------------
// Kernel 3: LN(out) over channels + output projection + gate.
__global__ void __launch_bounds__(256)
out_kernel(const float* __restrict__ x_t, const __bf16* __restrict__ wbz,
           const float* __restrict__ bz,
           const float* __restrict__ ln_w, const float* __restrict__ ln_b,
           const __bf16* __restrict__ g_nat, float* __restrict__ out) {
    __shared__ float  xs[64 * 129];
    __shared__ __bf16 xn[64 * CDIM];

    const int tid = threadIdx.x;
    const int p0  = blockIdx.x * 64;   // 64 consecutive pixels (within one i-row)

    // Gather transpose: for each channel, 64 consecutive pixels are contiguous.
#pragma unroll
    for (int it = 0; it < 32; ++it) {
        int flat = it * 256 + tid;                   // 0..8191
        int cc = flat >> 6, pl = flat & 63;
        xs[pl * 129 + cc] = x_t[(size_t)cc * NN + p0 + pl];
    }
    __syncthreads();

    if (tid < 64) {
        const float* row = xs + tid * 129;
        float s = 0.f, ss = 0.f;
#pragma unroll 8
        for (int c = 0; c < CDIM; ++c) { float v = row[c]; s += v; ss += v * v; }
        float mu  = s * (1.0f / CDIM);
        float var = ss * (1.0f / CDIM) - mu * mu;
        float rs  = rsqrtf(var + 1e-5f);
#pragma unroll 8
        for (int c = 0; c < CDIM; ++c) {
            float v = (row[c] - mu) * rs * ln_w[c] + ln_b[c];
            xn[tid * CDIM + c] = (__bf16)v;
        }
    }
    __syncthreads();

    const int wave = tid >> 5, lane = tid & 31;
    const int rb = wave & 3, cb0 = (wave >> 2) * 4;
    const int h = lane >> 4, n = lane & 15;
    const int pbase = p0 + rb * 16 + 8 * h;

    v8f acc[4];
    gemm4(xn, wbz, rb, cb0, lane, acc);

#pragma unroll
    for (int c = 0; c < 4; ++c) {
        int o = (cb0 + c) * 16 + n;
        float bzv = bz[o];
#pragma unroll
        for (int r = 0; r < 8; ++r) {
            int p = pbase + r;
            float g = (float)g_nat[(size_t)p * CDIM + o];
            out[(size_t)p * CDIM + o] = (acc[c][r] + bzv) * g;
        }
    }
}

// ---------------------------------------------------------------------------
extern "C" void kernel_launch(void* const* d_in, const int* in_sizes, int n_in,
                              void* d_out, int out_size, void* d_ws, size_t ws_size,
                              hipStream_t stream) {
    (void)in_sizes; (void)n_in; (void)out_size; (void)ws_size;

    const float* z        = (const float*)d_in[0];
    const float* mask     = (const float*)d_in[1];
    const float* ln_in_w  = (const float*)d_in[2];
    const float* ln_in_b  = (const float*)d_in[3];
    const float* w_ap     = (const float*)d_in[4];
    const float* b_ap     = (const float*)d_in[5];
    const float* w_ag     = (const float*)d_in[6];
    const float* b_ag     = (const float*)d_in[7];
    const float* w_bp     = (const float*)d_in[8];
    const float* b_bp     = (const float*)d_in[9];
    const float* w_bg     = (const float*)d_in[10];
    const float* b_bg     = (const float*)d_in[11];
    const float* w_g      = (const float*)d_in[12];
    const float* b_g      = (const float*)d_in[13];
    const float* w_z      = (const float*)d_in[14];
    const float* b_z      = (const float*)d_in[15];
    const float* ln_out_w = (const float*)d_in[16];
    const float* ln_out_b = (const float*)d_in[17];

    // Workspace layout (bytes):
    //   [0,        262144)  : 6 bf16 weight matrices (ap, ag, bp, bg, g, z)
    //   [262144,   +64MB )  : a  bf16 [C][N][N]
    //   [67371008, +64MB )  : b  bf16 [C][N][N]
    //   [134479872,+64MB )  : g  bf16 [P][C]
    //   [201588736,+128MB)  : x  fp32 [C][N][N]
    char* ws = (char*)d_ws;
    __bf16* wb    = (__bf16*)(ws);
    __bf16* a_t   = (__bf16*)(ws + 262144);
    __bf16* b_t   = (__bf16*)(ws + 67371008);
    __bf16* g_nat = (__bf16*)(ws + 134479872);
    float*  x_t   = (float*) (ws + 201588736);

    cvt_kernel<<<384, 256, 0, stream>>>(w_ap, w_ag, w_bp, w_bg, w_g, w_z, wb);

    proj_kernel<<<NN / 64, 256, 0, stream>>>(z, mask, ln_in_w, ln_in_b, wb,
                                             b_ap, b_ag, b_bp, b_bg, b_g,
                                             a_t, b_t, g_nat);

    einsum_kernel<<<dim3(8, 8, 128), 256, 16384, stream>>>(a_t, b_t, x_t);

    out_kernel<<<NN / 64, 256, 0, stream>>>(x_t, wb + 5 * 16384, b_z,
                                            ln_out_w, ln_out_b, g_nat,
                                            (float*)d_out);
}
